// RQVAE_17068200035054
// MI455X (gfx1250) — compile-verified, matrix operand order checked by reading, not run on hardware
//
#include <hip/hip_runtime.h>
#include <hip/hip_bf16.h>
#include <math.h>

// ---------------------------------------------------------------------------
// RQ-VAE forward for gfx1250 (MI455X, wave32).
// All matrix work (implicit-GEMM convs + codebook distance search) runs on
// v_wmma_f32_16x16x32_f16 with fp32 accumulation. Upsample2x is fused into
// the conv address math so no 2x-resolution tensor is ever materialized.
// Conv blocks compute a 64x64 output tile (4 waves x 4 WMMA accumulators)
// with the 64x32 f16 weight tile staged in LDS; the RQ argmin stages each
// 16-code tile in LDS shared by all 4 waves. Workspace ~51 MB, L2-resident.
// ---------------------------------------------------------------------------

typedef __attribute__((ext_vector_type(16))) _Float16 v16h;
typedef __attribute__((ext_vector_type(8)))  float    v8f;

// K-slot inside a 16x32 f16 A/B fragment for element j of lane `lane`.
// (16-bit A 16x32 layout: lanes 0-15 hold K 0..7 / 16..23, lanes 16-31 hold
//  K 8..15 / 24..31; B uses the mirrored mapping with N across lanes.)
__device__ __forceinline__ int kloc(int lane, int j) {
  return ((j >> 3) << 4) + (((lane >> 4) & 1) << 3) + (j & 7);
}

// ---------------------------------------------------------------------------
// Implicit-GEMM conv: M=Cout, N=B*Ho*Wo, K=Cin*ksz*ksz.
// Block = 128 threads = 4 waves. Block tile = 64(M) x 64(N):
//   - 64x32 f16 weight tile staged in LDS, shared by all waves
//   - each wave owns a 16-col N tile, loads its B fragment once per K step,
//     and reuses it across 4 WMMAs (4 M sub-tiles -> 4 accumulators).
// ups=1 reads the virtual 2x-upsampled input (Ho=2H, Wo=2W, stride=1).
// ---------------------------------------------------------------------------
__global__ __launch_bounds__(128) void conv_wmma(
    const float* __restrict__ X, const float* __restrict__ Wt,
    const float* __restrict__ Bs, float* __restrict__ Y,
    int Btch, int Cin, int H, int W, int Cout, int Ho, int Wo,
    int ksz, int stride, int pad, int ups)
{
  __shared__ _Float16 As[64 * 32];      // 4 KB weight tile
  const int lane  = threadIdx.x & 31;
  const int wv    = threadIdx.x >> 5;
  const int Kdim  = Cin * ksz * ksz;
  const int Ntot  = Btch * Ho * Wo;
  const int mtile = blockIdx.y << 6;    // 64 output channels per block
  const int ntile = (blockIdx.x << 6) + (wv << 4);

  const int ncol = ntile + (lane & 15);
  int bi = 0, oy = 0, ox = 0;
  if (ncol < Ntot) {
    bi = ncol / (Ho * Wo);
    int rem = ncol - bi * (Ho * Wo);
    oy = rem / Wo;
    ox = rem - oy * Wo;
  }

  v8f acc[4];
#pragma unroll
  for (int s = 0; s < 4; ++s) acc[s] = (v8f){};

  for (int k0 = 0; k0 < Kdim; k0 += 32) {
    // ---- prefetch next weight K-tile into caches (global_prefetch_b8) ----
    if (k0 + 32 < Kdim) {
      int pm = mtile + ((int)threadIdx.x >> 1);
      if (pm < Cout)
        __builtin_prefetch(&Wt[pm * Kdim + k0 + 32 + ((threadIdx.x & 1) << 4)],
                           0, 1);
    }
    // ---- stage weight tile A[64][32] in LDS (shared by all 4 waves) ----
    __syncthreads();
#pragma unroll
    for (int t = 0; t < 16; ++t) {
      int idx = ((int)threadIdx.x << 4) + t;  // 0..2047
      int m  = idx >> 5;
      int kk = idx & 31;
      int gm = mtile + m;
      int gk = k0 + kk;
      float v = 0.0f;
      if (gm < Cout && gk < Kdim) v = Wt[gm * Kdim + gk];
      As[idx] = (_Float16)v;
    }
    __syncthreads();

    // ---- per-wave im2col B fragment (loaded once, reused by 4 WMMAs) ----
    v16h b;
#pragma unroll
    for (int j = 0; j < 16; ++j) {
      const int kk = kloc(lane, j);
      float bv = 0.0f;
      const int gk = k0 + kk;
      if (ncol < Ntot && gk < Kdim) {
        int cin = gk / (ksz * ksz);
        int rc  = gk - cin * (ksz * ksz);
        int r   = rc / ksz;
        int c   = rc - r * ksz;
        int iy  = oy * stride + r - pad;
        int ix  = ox * stride + c - pad;
        if (ups) {
          if (iy >= 0 && iy < (H << 1) && ix >= 0 && ix < (W << 1))
            bv = X[((bi * Cin + cin) * H + (iy >> 1)) * W + (ix >> 1)];
        } else {
          if (iy >= 0 && iy < H && ix >= 0 && ix < W)
            bv = X[((bi * Cin + cin) * H + iy) * W + ix];
        }
      }
      b[j] = (_Float16)bv;
    }

    // ---- 4 WMMAs: one per 16-row M sub-tile ----
#pragma unroll
    for (int s = 0; s < 4; ++s) {
      v16h a;
#pragma unroll
      for (int j = 0; j < 16; ++j)
        a[j] = As[(((s << 4) + (lane & 15)) << 5) + kloc(lane, j)];
      acc[s] = __builtin_amdgcn_wmma_f32_16x16x32_f16(
          false, a, false, b, (short)0, acc[s], false, false);
    }
  }

  // ---- epilogue: bias + store (C/D layout: M = v + 8*(lane>=16), N = lane&15)
#pragma unroll
  for (int s = 0; s < 4; ++s) {
    const int mbase = mtile + (s << 4) + (((lane >> 4) & 1) << 3);
#pragma unroll
    for (int v = 0; v < 8; ++v) {
      int m = mbase + v;
      if (m < Cout && ncol < Ntot)
        Y[((bi * Cout + m) * Ho + oy) * Wo + ox] = acc[s][v] + Bs[m];
    }
  }
}

// ---------------------------------------------------------------------------
// GroupNorm (8 groups) with optional fused SiLU / residual add.
// One block per (batch, group); one-pass sum/sum-of-squares mean/var with LDS tree reduction.
// ---------------------------------------------------------------------------
__global__ __launch_bounds__(256) void groupnorm_k(
    const float* __restrict__ X, const float* __restrict__ Res,
    const float* __restrict__ Gm, const float* __restrict__ Bt,
    float* __restrict__ Y, int C, int H, int W, int doSilu, int doAdd)
{
  const int b  = blockIdx.x >> 3;
  const int g  = blockIdx.x & 7;
  const int Cg = C >> 3;
  const int cnt = Cg * H * W;
  const long base = ((long)b * C + (long)g * Cg) * (long)(H * W);

  __shared__ float s1[256], s2[256];
  float sum = 0.0f, ss = 0.0f;
  for (int i = threadIdx.x; i < cnt; i += 256) {
    float v = X[base + i];
    sum += v; ss += v * v;
  }
  s1[threadIdx.x] = sum; s2[threadIdx.x] = ss;
  __syncthreads();
  for (int o = 128; o > 0; o >>= 1) {
    if ((int)threadIdx.x < o) {
      s1[threadIdx.x] += s1[threadIdx.x + o];
      s2[threadIdx.x] += s2[threadIdx.x + o];
    }
    __syncthreads();
  }
  const float inv  = 1.0f / (float)cnt;
  const float mean = s1[0] * inv;
  const float var  = s2[0] * inv - mean * mean;
  const float rstd = rsqrtf(var + 1e-5f);

  for (int i = threadIdx.x; i < cnt; i += 256) {
    int ch = g * Cg + i / (H * W);
    float v = (X[base + i] - mean) * rstd * Gm[ch] + Bt[ch];
    if (doSilu) v = v / (1.0f + __expf(-v));
    if (doAdd)  v += Res[base + i];
    Y[base + i] = v;
  }
}

// ---------------------------------------------------------------------------
// Codebook row norms: CN[k] = sum_c CB[k][c]^2   (K=8192, C=256)
// ---------------------------------------------------------------------------
__global__ void cn_k(const float* __restrict__ CB, float* __restrict__ CN, int ncodes)
{
  int k = blockIdx.x * 256 + threadIdx.x;
  if (k < ncodes) {
    float s = 0.0f;
    for (int c = 0; c < 256; ++c) { float v = CB[k * 256 + c]; s += v * v; }
    CN[k] = s;
  }
}

// ---------------------------------------------------------------------------
// RQ nearest-code search. Z is residual (4,256,8,8) -> 256 rows of dim 256.
// Each of the 4 waves in a block owns 16 rows (A fragments resident in 64
// VGPRs). The block loops over 512 tiles of 16 codes; each tile (16x256 f16,
// 8 KB) is staged once in LDS and consumed by all 4 waves (8 WMMAs each).
// d2 = |z|^2 - 2 z.c + |c|^2 with running per-element argmin, then a
// shfl_xor min-reduction across the 16-lane half that holds each row.
// 4 blocks x 4 waves = 16 waves = 256 rows.
// ---------------------------------------------------------------------------
__global__ __launch_bounds__(128) void rq_argmin_k(
    const float* __restrict__ Z, const float* __restrict__ CB,
    const float* __restrict__ CN, int* __restrict__ Idx)
{
  __shared__ _Float16 CBs[16 * 256];    // 8 KB code tile
  const int lane  = threadIdx.x & 31;
  const int wv    = threadIdx.x >> 5;
  const int mtile = ((blockIdx.x << 2) + wv) << 4;
  const int half  = (lane >> 4) & 1;

  const int myrow = mtile + (lane & 15);
  const int zb = myrow >> 6;      // batch
  const int zp = myrow & 63;      // h*8+w

  // |z|^2 for my row, then broadcast the 8 values each half needs
  float zz = 0.0f;
  for (int c = 0; c < 256; ++c) {
    float v = Z[(zb * 256 + c) * 64 + zp];
    zz += v * v;
  }
  float zzv[8];
#pragma unroll
  for (int v = 0; v < 8; ++v)
    zzv[v] = __shfl(zz, v + half * 8, 32);

  // Preload entire A fragment set (z rows, K=256 -> 8 fragments, 64 VGPRs)
  v16h areg[8];
#pragma unroll
  for (int kb = 0; kb < 8; ++kb) {
#pragma unroll
    for (int j = 0; j < 16; ++j) {
      int c = kb * 32 + kloc(lane, j);
      areg[kb][j] = (_Float16)Z[(zb * 256 + c) * 64 + zp];
    }
  }

  float best[8]; int bidx[8];
#pragma unroll
  for (int v = 0; v < 8; ++v) { best[v] = 3.4e38f; bidx[v] = 0; }

  const int col = lane & 15;
  for (int ct = 0; ct < 512; ++ct) {
    // ---- stage 16-code tile in LDS, shared by all 4 waves ----
    __syncthreads();
#pragma unroll
    for (int t = 0; t < 32; ++t) {
      int idx = ((int)threadIdx.x << 5) + t;   // 0..4095
      int cd  = idx >> 8;                      // code within tile
      int c   = idx & 255;
      CBs[idx] = (_Float16)CB[((ct << 4) + cd) * 256 + c];
    }
    __syncthreads();

    const int code = (ct << 4) + col;
    v8f acc = {};
#pragma unroll
    for (int kb = 0; kb < 8; ++kb) {
      v16h bfrag;
#pragma unroll
      for (int j = 0; j < 16; ++j)
        bfrag[j] = CBs[(col << 8) + kb * 32 + kloc(lane, j)];
      acc = __builtin_amdgcn_wmma_f32_16x16x32_f16(
          false, areg[kb], false, bfrag, (short)0, acc, false, false);
    }
    const float cc = CN[code];
#pragma unroll
    for (int v = 0; v < 8; ++v) {
      float d2 = zzv[v] - 2.0f * acc[v] + cc;
      if (d2 < best[v]) { best[v] = d2; bidx[v] = code; }
    }
  }

  // min-reduce across the 16 lanes of each half (masks 1..8 stay in-half)
#pragma unroll
  for (int off = 1; off <= 8; off <<= 1) {
#pragma unroll
    for (int v = 0; v < 8; ++v) {
      float ob = __shfl_xor(best[v], off, 32);
      int   oi = __shfl_xor(bidx[v], off, 32);
      if (ob < best[v] || (ob == best[v] && oi < bidx[v])) {
        best[v] = ob; bidx[v] = oi;
      }
    }
  }
  if ((lane & 15) == 0) {
#pragma unroll
    for (int v = 0; v < 8; ++v)
      Idx[mtile + half * 8 + v] = bidx[v];
  }
}

// ---------------------------------------------------------------------------
// Gather q = CB[idx], update quantized_sum, emit quantized_list slice,
// indices (as float) and commit-loss contribution (block-reduced atomics).
// tid enumerates the 65536 elements of (4,256,8,8).
// ---------------------------------------------------------------------------
__global__ __launch_bounds__(256) void rq_gather_k(
    const float* __restrict__ Zres, const int* __restrict__ Idx,
    const float* __restrict__ CB, float* __restrict__ Qsum,
    float* __restrict__ Qlist, float* __restrict__ IdxF,
    float* __restrict__ Loss)
{
  __shared__ float sl[256];
  int tid = blockIdx.x * 256 + threadIdx.x;
  float d2 = 0.0f;
  if (tid < 65536) {
    int b   = tid >> 14;        // / (256*64)
    int r   = tid & 16383;
    int c   = r >> 6;
    int pos = r & 63;
    int n   = (b << 6) + pos;
    int code = Idx[n];
    float q = CB[code * 256 + c];
    Qsum[tid] += q;
    Qlist[tid] = q;
    if (c == 0) IdxF[n] = (float)code;
    float diff = Zres[tid] - q;
    d2 = diff * diff;
  }
  sl[threadIdx.x] = d2;
  __syncthreads();
  for (int o = 128; o > 0; o >>= 1) {
    if ((int)threadIdx.x < o) sl[threadIdx.x] += sl[threadIdx.x + o];
    __syncthreads();
  }
  if (threadIdx.x == 0) atomicAdd(Loss, sl[0] * (0.25f / 65536.0f));
}

// ---------------------------------------------------------------------------
// small elementwise helpers
// ---------------------------------------------------------------------------
__global__ void sub_k(const float* __restrict__ A, const float* __restrict__ B,
                      float* __restrict__ O, int n)
{ int i = blockIdx.x * 256 + threadIdx.x; if (i < n) O[i] = A[i] - B[i]; }

__global__ void zero_k(float* __restrict__ p, int n)
{ int i = blockIdx.x * 256 + threadIdx.x; if (i < n) p[i] = 0.0f; }

__global__ void copy_k(const float* __restrict__ a, float* __restrict__ o, int n)
{ int i = blockIdx.x * 256 + threadIdx.x; if (i < n) o[i] = a[i]; }

__global__ void tanh_k(float* __restrict__ p, int n)
{ int i = blockIdx.x * 256 + threadIdx.x; if (i < n) p[i] = tanhf(p[i]); }

// ---------------------------------------------------------------------------
// Host orchestration.
// d_in layout follows JAX's sorted-dict pytree flatten of setup_inputs():
//   0: x
//   1..4: codebooks[0..3]
//   5..76: dec blocks (18 each: res0{b1,b2,bt1,bt2,g1,g2,w1,w2}, res1{...},
//                      up{b,w})
//   77,78: dec_out {b,w}
//   79..150: enc blocks (18 each: down{b,w}, res0{...}, res1{...})
//   151,152: enc_to_latent {b,w}
//   153,154: latent_to_dec {b,w}
// d_out: recon(196608) | qsum(65536) | idx as float(1024) | qlist(262144) |
//        commit loss(1)
// ---------------------------------------------------------------------------
extern "C" void kernel_launch(void* const* d_in, const int* in_sizes, int n_in,
                              void* d_out, int out_size, void* d_ws, size_t ws_size,
                              hipStream_t stream)
{
  (void)in_sizes; (void)n_in; (void)out_size; (void)ws_size;

  const float* x = (const float*)d_in[0];
  auto F = [&](int i) { return (const float*)d_in[i]; };

  float* ws   = (float*)d_ws;
  float* bufA = ws;                       // 4*256*64*64 = 4194304 floats each
  float* bufB = ws + 4194304;
  float* bufC = ws + 8388608;
  float* Z    = ws + 12582912;            // 65536
  float* QSUM = Z + 65536;                // 65536
  float* ZRES = QSUM + 65536;             // 65536
  float* CN   = ZRES + 65536;             // 4*8192
  int*   IDX  = (int*)(CN + 32768);       // 256

  float* out       = (float*)d_out;
  float* out_recon = out;                         // 196608
  float* out_qsum  = out + 196608;                // 65536
  float* out_idx   = out + 196608 + 65536;        // 1024
  float* out_qlist = out_idx + 1024;              // 262144
  float* out_loss  = out_qlist + 262144;          // 1

  auto conv = [&](const float* Xp, const float* Wp, const float* Bp, float* Yp,
                  int Cin, int H, int W, int Cout, int Ho, int Wo,
                  int ksz, int stride, int pad, int ups) {
    int Ntot = 4 * Ho * Wo;
    dim3 grid((Ntot + 63) / 64, (Cout + 63) / 64);
    conv_wmma<<<grid, 128, 0, stream>>>(Xp, Wp, Bp, Yp, 4, Cin, H, W,
                                        Cout, Ho, Wo, ksz, stride, pad, ups);
  };
  auto gnorm = [&](const float* Xp, const float* Rp, const float* Gp,
                   const float* Bp, float* Yp, int C, int H, int W,
                   int silu, int add) {
    groupnorm_k<<<dim3(4 * 8), 256, 0, stream>>>(Xp, Rp, Gp, Bp, Yp,
                                                 C, H, W, silu, add);
  };
  // rb = base index of {b1,b2,bt1,bt2,g1,g2,w1,w2}
  auto resblock = [&](float* cur, float* t1, float* t2, int rb,
                      int dim, int H, int W) {
    conv(cur, F(rb + 6), F(rb + 0), t1, dim, H, W, dim, H, W, 3, 1, 1, 0);
    gnorm(t1, nullptr, F(rb + 4), F(rb + 2), t1, dim, H, W, 1, 0);   // GN+SiLU
    conv(t1, F(rb + 7), F(rb + 1), t2, dim, H, W, dim, H, W, 3, 1, 1, 0);
    gnorm(t2, cur, F(rb + 5), F(rb + 3), cur, dim, H, W, 0, 1);      // GN+res
  };

  // ------------------------------ encoder ------------------------------
  static const int CHSa[5] = {3, 256, 512, 512, 1024};
  const float* curIn = x;
  int H = 128;
  for (int blk = 0; blk < 4; ++blk) {
    const int base = 79 + blk * 18;     // down{b,w}, res0, res1
    const int Cin = CHSa[blk], Cout = CHSa[blk + 1];
    const int Ho = H >> 1;
    float *dst, *t1, *t2;
    if (curIn == bufA) { dst = bufB; t1 = bufC; t2 = bufA; }
    else               { dst = bufA; t1 = bufB; t2 = bufC; }
    conv(curIn, F(base + 1), F(base + 0), dst, Cin, H, H, Cout, Ho, Ho, 3, 2, 1, 0);
    resblock(dst, t1, t2, base + 2,  Cout, Ho, Ho);
    resblock(dst, t1, t2, base + 10, Cout, Ho, Ho);
    curIn = dst; H = Ho;
  }
  // enc_to_latent 1x1: 1024 -> 256 @ 8x8
  conv(curIn, F(152), F(151), Z, 1024, 8, 8, 256, 8, 8, 1, 1, 0, 0);

  // --------------------------- residual quantize ---------------------------
  zero_k<<<(65536 + 255) / 256, 256, 0, stream>>>(QSUM, 65536);
  zero_k<<<1, 256, 0, stream>>>(out_loss, 1);
  for (int l = 0; l < 4; ++l)
    cn_k<<<(8192 + 255) / 256, 256, 0, stream>>>(F(1 + l), CN + l * 8192, 8192);

  for (int l = 0; l < 4; ++l) {
    sub_k<<<256, 256, 0, stream>>>(Z, QSUM, ZRES, 65536);
    rq_argmin_k<<<4, 128, 0, stream>>>(ZRES, F(1 + l), CN + l * 8192, IDX);
    rq_gather_k<<<256, 256, 0, stream>>>(ZRES, IDX, F(1 + l), QSUM,
                                         out_qlist + l * 65536,
                                         out_idx + l * 256, out_loss);
  }
  copy_k<<<256, 256, 0, stream>>>(QSUM, out_qsum, 65536);

  // ------------------------------ decoder ------------------------------
  // latent_to_dec 1x1: 256 -> 1024 @ 8x8
  conv(QSUM, F(154), F(153), bufA, 256, 8, 8, 1024, 8, 8, 1, 1, 0, 0);
  float* dcur = bufA;
  int Hd = 8;
  static const int ddim[4] = {1024, 512, 512, 256};
  static const int dup[4]  = {512, 512, 256, 3};
  for (int b = 0; b < 4; ++b) {
    const int base = 5 + b * 18;        // res0, res1, up{b,w}
    float *t1, *t2, *dst;
    if (dcur == bufA) { t1 = bufB; t2 = bufC; dst = bufB; }
    else              { t1 = bufA; t2 = bufC; dst = bufA; }
    resblock(dcur, t1, t2, base + 0, ddim[b], Hd, Hd);
    resblock(dcur, t1, t2, base + 8, ddim[b], Hd, Hd);
    // fused upsample2x + 3x3 conv (reads virtual 2H x 2W image)
    conv(dcur, F(base + 17), F(base + 16), dst, ddim[b], Hd, Hd,
         dup[b], Hd * 2, Hd * 2, 3, 1, 1, 1);
    dcur = dst; Hd *= 2;
  }
  // dec_out 3x3: 3 -> 3 @ 128, then tanh
  conv(dcur, F(78), F(77), out_recon, 3, 128, 128, 3, 128, 128, 3, 1, 1, 0);
  tanh_k<<<(196608 + 255) / 256, 256, 0, stream>>>(out_recon, 196608);
}
